// SlidingWindowAttention_12369505812959
// MI455X (gfx1250) — matrix operand order repaired and checked by reading, not compiled, not run on hardware
//
#include <hip/hip_runtime.h>
#include <hip/hip_bf16.h>

// ---------------- problem constants ----------------
#define S_LEN 4096
#define HIDN  4096
#define NH    32
#define NKV   8
#define GQ    4          // NH / NKV
#define DH    128
#define WINW  512
#define SCALEF 0.08838834764831845f  // 128^-0.5
#define EPSV  1e-6f

// ---------------- tiling ----------------
#define BM 128
#define BN 128
#define BK 32
#define LDA (BK + 8)    // 40 elems = 80B rows -> every 16B run aligned
#define LDC (BN + 4)    // fp32 stage stride (528B, 16B aligned)
#define LDD (DH + 8)    // 136 elems = 272B rows (16B aligned)

typedef __bf16 bf16_t;
typedef __bf16 bf16x16 __attribute__((ext_vector_type(16)));
typedef float  f32x8   __attribute__((ext_vector_type(8)));
typedef int    v4i     __attribute__((vector_size(16)));  // matches builtin proto

#define AS3 __attribute__((address_space(3)))
#define AS1 __attribute__((address_space(1)))

#if __has_builtin(__builtin_amdgcn_global_load_async_to_lds_b128)
#define HAVE_ASYNC_LDS 1
#else
#define HAVE_ASYNC_LDS 0
#endif

__device__ __forceinline__ f32x8 wmma_bf16(bf16x16 a, bf16x16 b, f32x8 c) {
    return __builtin_amdgcn_wmma_f32_16x16x32_bf16(
        false, a, false, b, (short)0, c, false, false);
}

// pack two fp32 -> packed bf16 pair (round-to-nearest-even)
__device__ __forceinline__ unsigned bfpack2(float a, float b) {
    unsigned ua = __float_as_uint(a), ub = __float_as_uint(b);
    ua = ua + 0x7FFFu + ((ua >> 16) & 1u);
    ub = ub + 0x7FFFu + ((ub >> 16) & 1u);
    return (ua >> 16) | (ub & 0xFFFF0000u);
}

union frag_cv { uint4 q[2]; bf16x16 f; };

// A fragment: 16x32 bf16 from [M][K]-major LDS (row stride ldm elems).
// lane(hl,hi): row=row0+hl, halves 0..7 = K hi*8+0..7, halves 8..15 = +16.
__device__ __forceinline__ bf16x16 load_a_frag(const bf16_t* lds, int ldm,
                                               int row0, int k0) {
    const int lane = threadIdx.x & 31;
    const int hl = lane & 15, hi = lane >> 4;
    const bf16_t* p = lds + (size_t)(row0 + hl) * ldm + k0 + hi * 8;
    frag_cv cv;
    cv.q[0] = *(const uint4*)(p);
    cv.q[1] = *(const uint4*)(p + 16);
    return cv.f;
}

// B fragment: 32x16 bf16 from [N][K]-major LDS (row stride ldk elems).
// lane(hl,hi): col=col0+hl, halves 0..7 = K hi*8+0..7, halves 8..15 = +16.
__device__ __forceinline__ bf16x16 load_bt_frag(const bf16_t* lds, int ldk,
                                                int col0, int k0) {
    const int lane = threadIdx.x & 31;
    const int hl = lane & 15, hi = lane >> 4;
    const bf16_t* p = lds + (size_t)(col0 + hl) * ldk + k0 + hi * 8;
    frag_cv cv;
    cv.q[0] = *(const uint4*)(p);
    cv.q[1] = *(const uint4*)(p + 16);
    return cv.f;
}

__device__ __forceinline__ float hmax16(float v) {
#pragma unroll
    for (int m = 1; m < 16; m <<= 1) v = fmaxf(v, __shfl_xor(v, m, 32));
    return v;
}
__device__ __forceinline__ float hsum16(float v) {
#pragma unroll
    for (int m = 1; m < 16; m <<= 1) v += __shfl_xor(v, m, 32);
    return v;
}

// 16B global->LDS copy; async on CDNA5 when available
__device__ __forceinline__ void copy16(const bf16_t* g, bf16_t* l) {
#if HAVE_ASYNC_LDS
    __builtin_amdgcn_global_load_async_to_lds_b128(
        (AS1 v4i*)g, (AS3 v4i*)l, 0, 0);
#else
    *(uint4*)l = *(const uint4*)g;
#endif
}
__device__ __forceinline__ void copy_wait_async() {
#if HAVE_ASYNC_LDS
#if __has_builtin(__builtin_amdgcn_s_wait_asynccnt)
    __builtin_amdgcn_s_wait_asynccnt(0);
#else
    asm volatile("s_wait_asynccnt 0" ::: "memory");
#endif
#endif
}

// =====================================================================
// GEMM  Y = A(4096 x 4096) * W(4096 x N), bf16 WMMA, fp32 accumulate.
// MODE 0: Q proj -> RMSNorm+RoPE -> bf16 q [h][s][d]
// MODE 1: K proj -> RMSNorm+RoPE -> bf16 k [hkv][s][d]
// MODE 2: V proj ->                bf16 v^T [hkv][d][s]
// MODE 3: attn @ Wo (A is bf16) -> fp32 out [s][hid]
// =====================================================================
template <int MODE>
__global__ __launch_bounds__(256) void swa_gemm(
    const float* __restrict__ Af32, const bf16_t* __restrict__ Abf,
    const float* __restrict__ W, int N,
    const float* __restrict__ cosp, const float* __restrict__ sinp,
    const float* __restrict__ normw,
    bf16_t* __restrict__ obf, float* __restrict__ of32) {
    __shared__ __align__(16) bf16_t As[BM][LDA];  // [m][k]
    __shared__ __align__(16) bf16_t Bt[BN][LDA];  // [n][k]  fragment-major
    __shared__ __align__(16) float  Cs[BM][LDC];

    const int tid  = threadIdx.x;
    const int lane = tid & 31;
    const int wid  = tid >> 5;
    const int hl   = lane & 15, hi = lane >> 4;
    const int q0   = blockIdx.x * BM;
    const int n0b  = blockIdx.y * BN;
    const int m0   = (wid & 3) * 32;
    const int n0   = (wid >> 2) * 64;

    f32x8 acc[2][4];
#pragma unroll
    for (int i = 0; i < 2; ++i)
#pragma unroll
        for (int j = 0; j < 4; ++j) acc[i][j] = f32x8{0, 0, 0, 0, 0, 0, 0, 0};

    const int arow = tid >> 1, acol = (tid & 1) * 16;  // A tile 128x32
    const int brow = tid >> 3, bcol = (tid & 7) * 16;  // W tile 32x128

    for (int kt = 0; kt < HIDN; kt += BK) {
        // ---- stage A: fp32 -> packed bf16 (2 x b128 store) or bf16 copy ----
        if (MODE == 3) {
            const uint4* s4 =
                (const uint4*)(Abf + (size_t)(q0 + arow) * HIDN + kt + acol);
            *(uint4*)&As[arow][acol]     = s4[0];
            *(uint4*)&As[arow][acol + 8] = s4[1];
        } else {
            const float4* src =
                (const float4*)(Af32 + (size_t)(q0 + arow) * HIDN + kt + acol);
            float4 v0 = src[0], v1 = src[1], v2 = src[2], v3 = src[3];
            uint4 w0, w1;
            w0.x = bfpack2(v0.x, v0.y); w0.y = bfpack2(v0.z, v0.w);
            w0.z = bfpack2(v1.x, v1.y); w0.w = bfpack2(v1.z, v1.w);
            w1.x = bfpack2(v2.x, v2.y); w1.y = bfpack2(v2.z, v2.w);
            w1.z = bfpack2(v3.x, v3.y); w1.w = bfpack2(v3.z, v3.w);
            *(uint4*)&As[arow][acol]     = w0;
            *(uint4*)&As[arow][acol + 8] = w1;
        }
        // ---- stage W transposed into [n][k] fragment-major layout ----
        const float* bsrc = W + (size_t)(kt + brow) * N + n0b + bcol;
        if (kt + BK < HIDN)
            __builtin_prefetch(W + (size_t)(kt + BK + brow) * N + n0b + bcol, 0, 3);
#pragma unroll
        for (int i = 0; i < 16; i += 4) {
            float4 v = ((const float4*)bsrc)[i >> 2];
            Bt[bcol + i + 0][brow] = (bf16_t)v.x;
            Bt[bcol + i + 1][brow] = (bf16_t)v.y;
            Bt[bcol + i + 2][brow] = (bf16_t)v.z;
            Bt[bcol + i + 3][brow] = (bf16_t)v.w;
        }
        __syncthreads();

        // batch all fragment loads (distinct regs -> clause + single wait)
        bf16x16 a0 = load_a_frag(&As[0][0], LDA, m0, 0);
        bf16x16 a1 = load_a_frag(&As[0][0], LDA, m0 + 16, 0);
        bf16x16 bfr[4];
#pragma unroll
        for (int j = 0; j < 4; ++j)
            bfr[j] = load_bt_frag(&Bt[0][0], LDA, n0 + j * 16, 0);
#pragma unroll
        for (int j = 0; j < 4; ++j) {
            acc[0][j] = wmma_bf16(a0, bfr[j], acc[0][j]);
            acc[1][j] = wmma_bf16(a1, bfr[j], acc[1][j]);
        }
        __syncthreads();
    }

    // ---- spill accumulators to fp32 stage (C layout: row=r+hi*8, col=hl) ----
#pragma unroll
    for (int i = 0; i < 2; ++i)
#pragma unroll
        for (int j = 0; j < 4; ++j)
#pragma unroll
            for (int r = 0; r < 8; ++r)
                Cs[m0 + i * 16 + hi * 8 + r][n0 + j * 16 + hl] = acc[i][j][r];
    __syncthreads();

    // ---- epilogue: 2 threads per row, 64 cols each ----
    const int row = tid >> 1;
    const int c0  = (tid & 1) * 64;
    const int s   = q0 + row;

    if (MODE == 0 || MODE == 1) {
        float ss = 0.f;
#pragma unroll
        for (int d = 0; d < 64; ++d) {
            float x = Cs[row][c0 + d];
            ss += x * x;
        }
        ss += __shfl_xor(ss, 1, 32);
        const float rs = rsqrtf(ss * (1.0f / DH) + EPSV);
        const float* crow = cosp + (size_t)s * DH;
        const float* srow = sinp + (size_t)s * DH;
        unsigned obuf[32];
#pragma unroll 4
        for (int d0 = 0; d0 < 64; d0 += 2) {
            float o2[2];
#pragma unroll
            for (int u = 0; u < 2; ++u) {
                const int d  = c0 + d0 + u;
                const int dp = d ^ 64;
                float xn  = Cs[row][d]  * rs * normw[d];
                float xr  = Cs[row][dp] * rs * normw[dp];
                float rot = (d < 64) ? -xr : xr;  // rotate_half
                o2[u] = xn * crow[d] + rot * srow[d];
            }
            obuf[d0 >> 1] = bfpack2(o2[0], o2[1]);
        }
        uint4* dst = (uint4*)(obf + ((size_t)blockIdx.y * S_LEN + s) * DH + c0);
#pragma unroll
        for (int i = 0; i < 8; ++i) dst[i] = ((uint4*)obuf)[i];
    } else if (MODE == 2) {
        const int hkv = blockIdx.y;  // V stored transposed: [hkv][d][s]
        for (int d0 = 0; d0 < 64; ++d0) {
            const int d = c0 + d0;
            obf[((size_t)hkv * DH + d) * S_LEN + s] = (bf16_t)Cs[row][d];
        }
    } else {
        float4* dst = (float4*)(of32 + (size_t)s * HIDN + n0b + c0);
#pragma unroll
        for (int i = 0; i < 16; ++i) dst[i] = *(const float4*)&Cs[row][c0 + i * 4];
    }
}

// =====================================================================
// Sliding-window flash attention. One block = 128 queries x 1 head.
// Keys iterated in 128-wide blocks covering (s_q - 511, s_q].
// =====================================================================
__global__ __launch_bounds__(256) void swa_attn(
    const bf16_t* __restrict__ q, const bf16_t* __restrict__ k,
    const bf16_t* __restrict__ vt, bf16_t* __restrict__ o) {
    __shared__ __align__(16) bf16_t Qs[BM][DH];   // [m][d]   A-major
    __shared__ __align__(16) bf16_t Kn[BM][LDD];  // [n][d]   B-major for QK^T
    __shared__ __align__(16) bf16_t Vt[DH][LDD];  // [d][n]   B-major for PV
    __shared__ __align__(16) bf16_t Ps[BM][LDD];  // [m][n]   A-major probs

    const int tid  = threadIdx.x;
    const int lane = tid & 31;
    const int wid  = tid >> 5;  // 8 waves, wave owns 16 query rows
    const int hl   = lane & 15, hi = lane >> 4;
    const int q0   = blockIdx.x * BM;
    const int h    = blockIdx.y;
    const int hkv  = h / GQ;
    const int r2   = tid >> 1, c2 = (tid & 1) * 64;  // tile copy coords

    {   // load Q tile
        const uint4* s4 = (const uint4*)(q + ((size_t)h * S_LEN + q0 + r2) * DH + c2);
        uint4* dst = (uint4*)&Qs[r2][c2];
#pragma unroll
        for (int i = 0; i < 8; ++i) dst[i] = s4[i];
    }
    __syncthreads();

    bf16x16 aq[4];
#pragma unroll
    for (int kk = 0; kk < 4; ++kk)
        aq[kk] = load_a_frag(&Qs[0][0], DH, wid * 16, kk * 32);

    f32x8 oacc[8];
#pragma unroll
    for (int nt = 0; nt < 8; ++nt) oacc[nt] = f32x8{0, 0, 0, 0, 0, 0, 0, 0};
    float mrow[8], lrow[8];
#pragma unroll
    for (int r = 0; r < 8; ++r) { mrow[r] = -1e30f; lrow[r] = 0.f; }

    int lo = q0 - WINW + 1;
    if (lo < 0) lo = 0;
    const int kb0 = (lo / BM) * BM;

    for (int kb = kb0; kb < q0 + BM; kb += BM) {
        __syncthreads();
        {   // K rows (natural [s][d]) and V^T rows ([d][s]) -- both contiguous
            const bf16_t* kg = k + ((size_t)hkv * S_LEN + kb + r2) * DH + c2;
            const bf16_t* vg = vt + ((size_t)hkv * DH + r2) * S_LEN + kb + c2;
#pragma unroll
            for (int i = 0; i < 8; ++i) copy16(kg + i * 8, &Kn[r2][c2 + i * 8]);
#pragma unroll
            for (int i = 0; i < 8; ++i) copy16(vg + i * 8, &Vt[r2][c2 + i * 8]);
        }
        copy_wait_async();
        __syncthreads();

        // ---- S = Q K^T  (kk-outer: 8 independent accumulator chains) ----
        f32x8 sc[8];
#pragma unroll
        for (int nt = 0; nt < 8; ++nt) sc[nt] = f32x8{0, 0, 0, 0, 0, 0, 0, 0};
#pragma unroll
        for (int kk = 0; kk < 4; ++kk)
#pragma unroll
            for (int nt = 0; nt < 8; ++nt) {
                bf16x16 b = load_bt_frag(&Kn[0][0], LDD, nt * 16, kk * 32);
                sc[nt] = wmma_bf16(aq[kk], b, sc[nt]);
            }

        // ---- scale + sliding-window mask + row max ----
        float mcur[8];
#pragma unroll
        for (int r = 0; r < 8; ++r) mcur[r] = -1e30f;
#pragma unroll
        for (int nt = 0; nt < 8; ++nt) {
            const int sk = kb + nt * 16 + hl;
#pragma unroll
            for (int r = 0; r < 8; ++r) {
                const int sq = q0 + wid * 16 + hi * 8 + r;
                float val = sc[nt][r] * SCALEF;
                const bool ok = (sk <= sq) && (sk + WINW > sq);
                val = ok ? val : -1e30f;
                sc[nt][r] = val;
                mcur[r] = fmaxf(mcur[r], val);
            }
        }
#pragma unroll
        for (int r = 0; r < 8; ++r) mcur[r] = hmax16(mcur[r]);

        float alpha[8], mnew[8];
#pragma unroll
        for (int r = 0; r < 8; ++r) {
            mnew[r]  = fmaxf(mrow[r], mcur[r]);
            alpha[r] = __expf(mrow[r] - mnew[r]);
            mrow[r]  = mnew[r];
        }

        float lcur[8];
#pragma unroll
        for (int r = 0; r < 8; ++r) lcur[r] = 0.f;
#pragma unroll
        for (int nt = 0; nt < 8; ++nt)
#pragma unroll
            for (int r = 0; r < 8; ++r) {
                float p = __expf(sc[nt][r] - mnew[r]);
                sc[nt][r] = p;
                lcur[r] += p;
            }
#pragma unroll
        for (int r = 0; r < 8; ++r) {
            lcur[r] = hsum16(lcur[r]);
            lrow[r] = lrow[r] * alpha[r] + lcur[r];
        }
#pragma unroll
        for (int nt = 0; nt < 8; ++nt)
#pragma unroll
            for (int r = 0; r < 8; ++r) oacc[nt][r] *= alpha[r];

        // ---- P (C layout) -> LDS A-major; wave-private rows ----
#pragma unroll
        for (int nt = 0; nt < 8; ++nt)
#pragma unroll
            for (int r = 0; r < 8; ++r)
                Ps[wid * 16 + hi * 8 + r][nt * 16 + hl] = (bf16_t)sc[nt][r];
        asm volatile("s_wait_dscnt 0" ::: "memory");

        // ---- O += P V ----
#pragma unroll
        for (int kk = 0; kk < 4; ++kk) {
            bf16x16 ap = load_a_frag(&Ps[0][0], LDD, wid * 16, kk * 32);
#pragma unroll
            for (int nt = 0; nt < 8; ++nt) {
                bf16x16 bv = load_bt_frag(&Vt[0][0], LDD, nt * 16, kk * 32);
                oacc[nt] = wmma_bf16(ap, bv, oacc[nt]);
            }
        }
    }

    // ---- normalize and write [s][h*D + d] bf16 for the Wo GEMM ----
#pragma unroll
    for (int nt = 0; nt < 8; ++nt)
#pragma unroll
        for (int r = 0; r < 8; ++r) {
            const int sq = q0 + wid * 16 + hi * 8 + r;
            float val = oacc[nt][r] / lrow[r];
            o[(size_t)sq * HIDN + h * DH + nt * 16 + hl] = (bf16_t)val;
        }
}

// =====================================================================
extern "C" void kernel_launch(void* const* d_in, const int* in_sizes, int n_in,
                              void* d_out, int out_size, void* d_ws,
                              size_t ws_size, hipStream_t stream) {
    const float* hid  = (const float*)d_in[0];
    const float* cosp = (const float*)d_in[1];
    const float* sinp = (const float*)d_in[2];
    const float* wq   = (const float*)d_in[3];
    const float* wk   = (const float*)d_in[4];
    const float* wv   = (const float*)d_in[5];
    const float* wo   = (const float*)d_in[6];
    const float* qn   = (const float*)d_in[7];
    const float* kn   = (const float*)d_in[8];
    float* out = (float*)d_out;

    char* ws = (char*)d_ws;
    const size_t q_bytes  = (size_t)NH * S_LEN * DH * sizeof(bf16_t);   // 32 MB
    const size_t kv_bytes = (size_t)NKV * S_LEN * DH * sizeof(bf16_t);  //  8 MB
    bf16_t* qb  = (bf16_t*)(ws);
    bf16_t* kbp = (bf16_t*)(ws + q_bytes);
    bf16_t* vtb = (bf16_t*)(ws + q_bytes + kv_bytes);
    bf16_t* ab  = (bf16_t*)(ws + q_bytes + 2 * kv_bytes);               // 32 MB

    dim3 blk(256);
    swa_gemm<0><<<dim3(S_LEN / BM, NH), blk, 0, stream>>>(
        hid, nullptr, wq, NH * DH, cosp, sinp, qn, qb, nullptr);
    swa_gemm<1><<<dim3(S_LEN / BM, NKV), blk, 0, stream>>>(
        hid, nullptr, wk, NKV * DH, cosp, sinp, kn, kbp, nullptr);
    swa_gemm<2><<<dim3(S_LEN / BM, NKV), blk, 0, stream>>>(
        hid, nullptr, wv, NKV * DH, nullptr, nullptr, nullptr, vtb, nullptr);
    swa_attn<<<dim3(S_LEN / BM, NH), blk, 0, stream>>>(qb, kbp, vtb, ab);
    swa_gemm<3><<<dim3(S_LEN / BM, HIDN / BN), blk, 0, stream>>>(
        nullptr, ab, wo, HIDN, nullptr, nullptr, nullptr, nullptr, out);
}
